// MoE_15573551415344
// MI455X (gfx1250) — compile-verified, matrix operand order checked by reading, not run on hardware
//
#include <hip/hip_runtime.h>

#define D_MODEL 1024
#define HIDDEN  4096
#define NEXPERT 8
#define NTOK    4096   // 2 * 2048
#define M_TILE  32

typedef __bf16 bf16;
typedef __bf16 bf16x16 __attribute__((ext_vector_type(16)));
typedef __bf16 bf16x8  __attribute__((ext_vector_type(8)));
typedef float  f32x8   __attribute__((ext_vector_type(8)));
typedef float  f32x4   __attribute__((ext_vector_type(4)));

#define CONCAT16(lo, hi) \
  __builtin_shufflevector((lo), (hi), 0, 1, 2, 3, 4, 5, 6, 7, \
                          8, 9, 10, 11, 12, 13, 14, 15)

// ---------------------------------------------------------------------------
// One-pass fp32 -> bf16 weight conversion into workspace (memory-bound,
// ~400MB of traffic ~= 17us; removes all per-WMMA cvt VALU from the GEMMs).
// ---------------------------------------------------------------------------
__global__ __launch_bounds__(256) void convert_bf16_kernel(
    const float* __restrict__ src, bf16* __restrict__ dst, size_t n)
{
  size_t i0 = ((size_t)blockIdx.x * 256 + threadIdx.x) * 8;
  size_t stride = (size_t)gridDim.x * 256 * 8;
  for (size_t i = i0; i < n; i += stride) {
    f32x4 a = *(const f32x4*)(src + i);
    f32x4 b = *(const f32x4*)(src + i + 4);
    bf16x8 o;
#pragma unroll
    for (int j = 0; j < 4; ++j) { o[j] = (bf16)a[j]; o[j + 4] = (bf16)b[j]; }
    *(bf16x8*)(dst + i) = o;
  }
}

// ---------------------------------------------------------------------------
// Router: logits = x @ w_gate^T, top-2, softmax over the 2 logits, build
// per-expert token lists (token index + gate weight) via atomics.
// ---------------------------------------------------------------------------
__global__ __launch_bounds__(256) void router_kernel(
    const float* __restrict__ x, const float* __restrict__ w_gate,
    int* __restrict__ cnt, int* __restrict__ tok_list,
    float* __restrict__ gate_list)
{
  int n = blockIdx.x * blockDim.x + threadIdx.x;
  if (n >= NTOK) return;
  const float* xr = x + (size_t)n * D_MODEL;

  float acc[NEXPERT];
#pragma unroll
  for (int e = 0; e < NEXPERT; ++e) acc[e] = 0.f;
  for (int d = 0; d < D_MODEL; ++d) {
    float xv = xr[d];
#pragma unroll
    for (int e = 0; e < NEXPERT; ++e) acc[e] += xv * w_gate[e * D_MODEL + d];
  }

  int i0 = 0;
#pragma unroll
  for (int e = 1; e < NEXPERT; ++e) if (acc[e] > acc[i0]) i0 = e;
  int i1 = (i0 == 0) ? 1 : 0;
#pragma unroll
  for (int e = 0; e < NEXPERT; ++e) if (e != i0 && acc[e] > acc[i1]) i1 = e;

  float p1  = __expf(acc[i1] - acc[i0]);   // softmax over {top1, top2}
  float inv = 1.0f / (1.0f + p1);
  float g0 = inv, g1 = p1 * inv;

  int p = atomicAdd(&cnt[i0], 1);
  tok_list[i0 * NTOK + p] = n;  gate_list[i0 * NTOK + p] = g0;
  p = atomicAdd(&cnt[i1], 1);
  tok_list[i1 * NTOK + p] = n;  gate_list[i1 * NTOK + p] = g1;
}

// ---------------------------------------------------------------------------
// Fused expert FFN: one workgroup = (expert e, tile of 32 routed tokens).
// 256 threads = 8 wave32. Each weight B-fragment feeds TWO WMMAs (two
// 16-token row blocks) -> 2x arithmetic intensity on the weight stream.
// Hidden dim processed in chunks of 128:
//   phase 1: each wave computes a 32x16 slice of h = x_tile @ w_in[e]^T
//            (K = 1024 reduction, bf16 WMMA), stores bf16 into LDS.
//   phase 2: each wave owns 128 output dims; accumulates
//            y += h_chunk @ w_out[e][cols, chunk]^T into 16 WMMA C-tiles.
// Weights arrive pre-converted bf16: fragment chunks are pure b128 loads.
// Finalize: scale rows by gate, fp32 global atomic add into out.
// ---------------------------------------------------------------------------
__global__ __launch_bounds__(256) void moe_ffn_kernel(
    const float* __restrict__ x, const bf16* __restrict__ w_in,
    const bf16* __restrict__ w_out, const int* __restrict__ cnt,
    const int* __restrict__ tok_list, const float* __restrict__ gate_list,
    float* __restrict__ out)
{
  const int e    = blockIdx.y;
  const int tile = blockIdx.x;
  const int c    = cnt[e];
  if (tile * M_TILE >= c) return;          // uniform early exit (no WMMA yet)

  const int tid  = threadIdx.x;
  const int wave = tid >> 5;
  const int lane = tid & 31;
  const int half = lane >> 4;              // 0: lanes 0-15, 1: lanes 16-31
  const int lrow = lane & 15;              // M (phase-1 A) / N (B, C) index
  const int koff = half * 8;

  __shared__ bf16  x_lds[M_TILE][D_MODEL + 8];   // 32 x 1024 bf16 token tile
  __shared__ bf16  h_lds[M_TILE][128 + 8];       // 32 x 128 bf16 hidden chunk
  __shared__ int   tok_s[M_TILE];
  __shared__ float gate_s[M_TILE];

  if (tid < M_TILE) {
    int idx = tile * M_TILE + tid;
    if (idx < c) {
      tok_s[tid]  = tok_list[e * NTOK + idx];
      gate_s[tid] = gate_list[e * NTOK + idx];
    } else {                      // pad slots: token 0 with gate 0 (adds 0.0f)
      tok_s[tid]  = 0;
      gate_s[tid] = 0.0f;
    }
  }
  __syncthreads();

  // gather 32 token rows, fp32 -> bf16, into LDS (coalesced over columns)
  for (int idx = tid; idx < M_TILE * D_MODEL; idx += 256) {
    int r = idx >> 10, col = idx & (D_MODEL - 1);
    x_lds[r][col] = (bf16)x[(size_t)tok_s[r] * D_MODEL + col];
  }
  __syncthreads();

  const bf16* wi = w_in  + (size_t)e * HIDDEN * D_MODEL;
  const bf16* wo = w_out + (size_t)e * D_MODEL * HIDDEN;

  f32x8 yacc[2][8];                        // [token row block][col tile]
#pragma unroll
  for (int rb = 0; rb < 2; ++rb)
#pragma unroll
    for (int t = 0; t < 8; ++t) yacc[rb][t] = (f32x8)0.0f;

  const int d0 = wave * 128;               // this wave's output-column block

  for (int h0 = 0; h0 < HIDDEN; h0 += 128) {
    // ---- phase 1: 32x16 h slice for hidden cols [h0+16*wave, +16)
    f32x8 hacc0 = (f32x8)0.0f, hacc1 = (f32x8)0.0f;
    const bf16* wrow = wi + (size_t)(h0 + wave * 16 + lrow) * D_MODEL;
    for (int kc = 0; kc < D_MODEL; kc += 32) {
      __builtin_prefetch(wrow + kc + 128, 0, 3);   // next chunks of this row
      const bf16* wr = wrow + kc + koff;
      bf16x8 blo = *(const bf16x8*)(wr);
      bf16x8 bhi = *(const bf16x8*)(wr + 16);
      bf16x16 b = CONCAT16(blo, bhi);

      const bf16* xr0 = &x_lds[lrow][kc + koff];
      const bf16* xr1 = &x_lds[16 + lrow][kc + koff];
      bf16x16 a0 = CONCAT16(*(const bf16x8*)(xr0), *(const bf16x8*)(xr0 + 16));
      bf16x16 a1 = CONCAT16(*(const bf16x8*)(xr1), *(const bf16x8*)(xr1 + 16));

      hacc0 = __builtin_amdgcn_wmma_f32_16x16x32_bf16(false, a0, false, b,
                                                      (short)0, hacc0, false, false);
      hacc1 = __builtin_amdgcn_wmma_f32_16x16x32_bf16(false, a1, false, b,
                                                      (short)0, hacc1, false, false);
    }
    // C layout: VGPR r, lanes 0-15 -> M=r; lanes 16-31 -> M=r+8; N=lane%16
#pragma unroll
    for (int r = 0; r < 8; ++r) {
      h_lds[r + half * 8][wave * 16 + lrow]      = (bf16)hacc0[r];
      h_lds[16 + r + half * 8][wave * 16 + lrow] = (bf16)hacc1[r];
    }
    __syncthreads();

    // ---- phase 2: yacc += h[32x128] @ w_out[e][d0..d0+127][h0..h0+127]^T
#pragma unroll
    for (int t = 0; t < 8; ++t) {
      const bf16* worow = wo + (size_t)(d0 + t * 16 + lrow) * HIDDEN + h0;
      __builtin_prefetch(worow + 128, 0, 3);       // next h-chunk of this row
#pragma unroll
      for (int kk = 0; kk < 128; kk += 32) {
        const bf16* wr = worow + kk + koff;
        bf16x8 blo = *(const bf16x8*)(wr);
        bf16x8 bhi = *(const bf16x8*)(wr + 16);
        bf16x16 b = CONCAT16(blo, bhi);

        const bf16* hr0 = &h_lds[lrow][kk + koff];
        const bf16* hr1 = &h_lds[16 + lrow][kk + koff];
        bf16x16 a0 = CONCAT16(*(const bf16x8*)(hr0), *(const bf16x8*)(hr0 + 16));
        bf16x16 a1 = CONCAT16(*(const bf16x8*)(hr1), *(const bf16x8*)(hr1 + 16));

        yacc[0][t] = __builtin_amdgcn_wmma_f32_16x16x32_bf16(false, a0, false, b,
                                                             (short)0, yacc[0][t], false, false);
        yacc[1][t] = __builtin_amdgcn_wmma_f32_16x16x32_bf16(false, a1, false, b,
                                                             (short)0, yacc[1][t], false, false);
      }
    }
    __syncthreads();   // h_lds rewritten next chunk
  }

  // ---- finalize: gate scaling + combine across the token's 2 experts
#pragma unroll
  for (int rb = 0; rb < 2; ++rb) {
#pragma unroll
    for (int t = 0; t < 8; ++t) {
#pragma unroll
      for (int r = 0; r < 8; ++r) {
        int   mrow  = rb * 16 + r + half * 8;
        int   token = tok_s[mrow];
        float g     = gate_s[mrow];
        int   col   = d0 + t * 16 + lrow;
        __hip_atomic_fetch_add(&out[(size_t)token * D_MODEL + col],
                               g * yacc[rb][t][r],
                               __ATOMIC_RELAXED, __HIP_MEMORY_SCOPE_AGENT);
      }
    }
  }
}

// ---------------------------------------------------------------------------
extern "C" void kernel_launch(void* const* d_in, const int* in_sizes, int n_in,
                              void* d_out, int out_size, void* d_ws, size_t ws_size,
                              hipStream_t stream) {
  const float* x      = (const float*)d_in[0];
  const float* w_gate = (const float*)d_in[1];
  const float* w_in   = (const float*)d_in[2];
  const float* w_out  = (const float*)d_in[3];
  float* out = (float*)d_out;

  const size_t W_ELEMS = (size_t)NEXPERT * HIDDEN * D_MODEL;  // 33.55M each

  char*  ws        = (char*)d_ws;
  int*   cnt       = (int*)ws;                                   // 256 B
  int*   tok_list  = (int*)(ws + 256);                           // 128 KB
  float* gate_list = (float*)(ws + 256 + NEXPERT * NTOK * sizeof(int));
  bf16*  wi_bf     = (bf16*)(ws + 256 + 2 * NEXPERT * NTOK * sizeof(int));
  bf16*  wo_bf     = wi_bf + W_ELEMS;                            // +67.1 MB each

  (void)hipMemsetAsync(cnt, 0, 256, stream);
  (void)hipMemsetAsync(out, 0, (size_t)out_size * sizeof(float), stream);

  convert_bf16_kernel<<<8192, 256, 0, stream>>>(w_in,  wi_bf, W_ELEMS);
  convert_bf16_kernel<<<8192, 256, 0, stream>>>(w_out, wo_bf, W_ELEMS);

  router_kernel<<<NTOK / 256, 256, 0, stream>>>(x, w_gate, cnt, tok_list, gate_list);

  dim3 grid(NTOK / M_TILE, NEXPERT);
  moe_ffn_kernel<<<grid, 256, 0, stream>>>(x, wi_bf, wo_bf, cnt, tok_list,
                                           gate_list, out);
}
